// Mamba3Block_14791867367993
// MI455X (gfx1250) — compile-verified, matrix operand order checked by reading, not run on hardware
//
#include <hip/hip_runtime.h>
#include <math.h>

// ---------------------------------------------------------------------------
// Mamba3-style block for MI455X (gfx1250, wave32).
// GEMMs: v_wmma_f32_16x16x32_bf16, operands staged to LDS by the Tensor Data
// Mover (tensor_load_to_lds, TENSORcnt) with double buffering.
// ---------------------------------------------------------------------------

typedef __attribute__((ext_vector_type(16))) __bf16 v16bf;
typedef __attribute__((ext_vector_type(8)))  __bf16 v8bf;
typedef __attribute__((ext_vector_type(4)))  __bf16 v4bf;
typedef __attribute__((ext_vector_type(8)))  float  v8f;
typedef __attribute__((ext_vector_type(4)))  float  v4f;
typedef __attribute__((ext_vector_type(4)))  unsigned int v4u;
typedef __attribute__((ext_vector_type(8)))  int v8i;
typedef __attribute__((ext_vector_type(4)))  int v4i;

#define B_DIM     2
#define L_DIM     4096
#define D_MODEL   1024
#define D_INNER   2048
#define DT_RANK   64
#define ML        ((size_t)B_DIM * L_DIM)      // 8192 rows for all GEMMs

// GEMM tiling
#define BM 64
#define BN 128
#define BK 32
#define LDSK 40      // padded LDS row stride in bf16 (= 16 dwords data + 4 dwords pad)

// scan chunking
#define CH 128
#define NC (L_DIM / CH)    // 32

// ---------------------------------------------------------------------------
// TDM: issue one 2-D bf16 tile load (tile_dim0 = BK = 32 elems per row) into
// LDS with pad_interval = 16 dwords, pad_amount = 4 dwords  ->  LDS row
// stride = 40 bf16 = the fragment layout the ds_load_b128 reads expect.
// ---------------------------------------------------------------------------
__device__ __forceinline__ void tdm_load_tile(const __bf16* g, unsigned lds_off,
                                              unsigned rowLen, unsigned tileRows)
{
    unsigned long long ga = (unsigned long long)(uintptr_t)g;
    v4u g0;
    g0[0] = 1u;                                           // count=1 (valid D#)
    g0[1] = lds_off;                                      // lds_addr (bytes)
    g0[2] = (unsigned)ga;                                 // global_addr[31:0]
    g0[3] = (unsigned)((ga >> 32) & 0x1FFFFFFu) | (2u << 30);  // [56:32] | type=2
    v8i g1;
    // dw0: wg_mask=0 | data_size=1(2B)<<16 | pad_en<<20 | pad_interval=3(16dw)<<22
    //      | pad_amount=3(4dw)<<25
    g1[0] = (int)((1u << 16) | (1u << 20) | (3u << 22) | (3u << 25));
    unsigned td0 = rowLen;          // tensor_dim0 (elements)
    unsigned td1 = 1u << 20;        // tensor_dim1: large (no OOB in our tiling)
    g1[1] = (int)((td0 & 0xFFFFu) << 16);                         // dim0[15:0]
    g1[2] = (int)(((td0 >> 16) & 0xFFFFu) | ((td1 & 0xFFFFu) << 16));
    g1[3] = (int)(((td1 >> 16) & 0xFFFFu) | ((unsigned)BK << 16)); // tile_dim0=32
    g1[4] = (int)(tileRows & 0xFFFFu);                             // tile_dim1
    g1[5] = (int)rowLen;                                           // dim0_stride[31:0]
    g1[6] = 0;
    g1[7] = 0;
    v4i z4 = {0, 0, 0, 0};
    v8i z8 = {0, 0, 0, 0, 0, 0, 0, 0};
    // amdgpu-toolchain (clang-23) form: 6 args, extra int32x8 group before cpol
    __builtin_amdgcn_tensor_load_to_lds(g0, g1, z4, z4, z8, 0);
}

// ---------------------------------------------------------------------------
// WMMA GEMM:  OUT[M,N] = A0*op(W0) (+ A1*op(W1)) (+ ADD) ; bf16 in, fp32 acc.
//   TRANS_W=1 : W is (N,K) row-major (use W^T)  -> B tile staged by TDM
//   TRANS_W=0 : W is (K,N) row-major (A_real)   -> B tile staged manually
//   EPI: 0 plain, 1 +=ADD, 2 decay epilogue.  OUT16EN: also emit bf16 copy.
// 256 threads = 8 waves (2x4); wave computes 32x32; block tile 64x128.
// ---------------------------------------------------------------------------
template<bool TRANS_W, bool HAS_A2, int EPI, bool OUT16EN>
__global__ __launch_bounds__(256)
void wmma_gemm(const __bf16* __restrict__ A0, const __bf16* __restrict__ W0,
               const __bf16* __restrict__ A1, const __bf16* __restrict__ W1,
               const float* __restrict__ ADD,
               const float* __restrict__ BIAS, const float* __restrict__ ALOG,
               float* __restrict__ OUT, __bf16* __restrict__ OUT16,
               int N, int K)
{
    __shared__ __bf16 smem[2 * BM * LDSK + 2 * BN * LDSK];
    // scalar (non-aggregate) LDS address math only — aggregate initializers
    // holding addrspacecast(LDS) break ld.lld static initializers.
    const unsigned ldsBase = (unsigned)(uintptr_t)(__bf16*)smem;  // byte offset of smem
    const unsigned aBytes  = (unsigned)(BM * LDSK * 2);
    const unsigned bBytes  = (unsigned)(BN * LDSK * 2);

    const int tid   = threadIdx.x;
    const int wave  = tid >> 5;
    const int lane  = tid & 31;
    const int wm    = wave >> 2;          // 0..1
    const int wn    = wave & 3;           // 0..3
    const int lrow  = lane & 15;
    const int lhalf = lane >> 4;          // ISA K-interleave half select

    const int m0 = blockIdx.y * BM;
    const int n0 = blockIdx.x * BN;

    v8f acc[2][2] = {};

    const int nK = K / BK;
    const int nT = HAS_A2 ? 2 * nK : nK;

    auto stage = [&](int t, int buf) {
        const __bf16* Ap = A0;
        const __bf16* Wp = W0;
        int kt = t;
        if (HAS_A2 && t >= nK) { Ap = A1; Wp = W1; kt = t - nK; }
        const int k0 = kt * BK;
        if (wave == 0) {
            tdm_load_tile(Ap + (size_t)m0 * K + k0,
                          ldsBase + (unsigned)buf * aBytes, (unsigned)K, BM);
            if (TRANS_W)
                tdm_load_tile(Wp + (size_t)n0 * K + k0,
                              ldsBase + 2 * aBytes + (unsigned)buf * bBytes,
                              (unsigned)K, BN);
        }
        if (!TRANS_W) {
            // W is KxN: transpose 32x128 bf16 tile into B buffer as [n][k]
            __bf16* dst = smem + 2 * BM * LDSK + buf * (BN * LDSK);
            #pragma unroll
            for (int i = 0; i < 2; ++i) {
                int idx = tid + i * 256;          // over [32 k][16 n-octets]
                int k  = idx >> 4;
                int n8 = (idx & 15) * 8;
                v8bf v = *(const v8bf*)&Wp[(size_t)(k0 + k) * N + n0 + n8];
                #pragma unroll
                for (int j = 0; j < 8; ++j)
                    dst[(n8 + j) * LDSK + k] = v[j];
            }
        }
    };

    stage(0, 0);

    for (int t = 0; t < nT; ++t) {
        const int cur = t & 1;
        __builtin_amdgcn_s_wait_tensorcnt((short)0);   // tile t landed in LDS
        __syncthreads();                               // visible to all waves
        if (t + 1 < nT) stage(t + 1, cur ^ 1);         // overlap with compute

        const __bf16* cA = smem + cur * (BM * LDSK);
        const __bf16* cB = smem + 2 * BM * LDSK + cur * (BN * LDSK);

        // fragments per ISA 16-bit lane layout:
        // lane<16: K {0..7,16..23}; lane>=16: K {8..15,24..31}
        v16bf afrag[2], bfrag[2];
        #pragma unroll
        for (int tm = 0; tm < 2; ++tm) {
            int base = (wm * 32 + tm * 16 + lrow) * LDSK + lhalf * 8;
            v8bf lo = *(const v8bf*)&cA[base];
            v8bf hi = *(const v8bf*)&cA[base + 16];
            afrag[tm] = __builtin_shufflevector(lo, hi,
                0,1,2,3,4,5,6,7,8,9,10,11,12,13,14,15);
        }
        #pragma unroll
        for (int tn = 0; tn < 2; ++tn) {
            int base = (wn * 32 + tn * 16 + lrow) * LDSK + lhalf * 8;
            v8bf lo = *(const v8bf*)&cB[base];
            v8bf hi = *(const v8bf*)&cB[base + 16];
            bfrag[tn] = __builtin_shufflevector(lo, hi,
                0,1,2,3,4,5,6,7,8,9,10,11,12,13,14,15);
        }
        #pragma unroll
        for (int tm = 0; tm < 2; ++tm)
            #pragma unroll
            for (int tn = 0; tn < 2; ++tn)
                acc[tm][tn] = __builtin_amdgcn_wmma_f32_16x16x32_bf16(
                    false, afrag[tm], false, bfrag[tn],
                    (short)0, acc[tm][tn], false, false);
    }

    // epilogue: lane owns col (lane%16), rows lhalf*8 + r
    #pragma unroll
    for (int tm = 0; tm < 2; ++tm) {
        #pragma unroll
        for (int tn = 0; tn < 2; ++tn) {
            int col   = n0 + wn * 32 + tn * 16 + lrow;
            int mbase = m0 + wm * 32 + tm * 16 + lhalf * 8;
            #pragma unroll
            for (int r = 0; r < 8; ++r) {
                size_t o = (size_t)(mbase + r) * N + col;
                float v = acc[tm][tn][r];
                if (EPI == 1) v += ADD[o];
                if (EPI == 2) {
                    v += BIAS[col];
                    float sp = (v > 20.f) ? v : log1pf(expf(v));   // softplus
                    v = expf(-expf(ALOG[col]) * sp);               // decay
                }
                OUT[o] = v;
                if (OUT16EN) OUT16[o] = (__bf16)v;
            }
        }
    }
}

// ---------------------------------------------------------------------------
// fp32 -> bf16 convert, 4 elems/thread (b128 load, b64 store)
// ---------------------------------------------------------------------------
__global__ __launch_bounds__(256)
void cvt_bf16_kernel(const float* __restrict__ in, __bf16* __restrict__ out)
{
    size_t i4 = (size_t)blockIdx.x * 256 + threadIdx.x;
    v4f f = *(const v4f*)&in[i4 * 4];
    v4bf h;
    #pragma unroll
    for (int j = 0; j < 4; ++j) h[j] = (__bf16)f[j];
    *(v4bf*)&out[i4 * 4] = h;
}

// ---------------------------------------------------------------------------
// Depthwise causal conv (D_CONV=4) + bias + SiLU; fp32 + bf16 outputs.
// ---------------------------------------------------------------------------
__global__ __launch_bounds__(256)
void conv_silu_kernel(const float* __restrict__ xz,
                      const float* __restrict__ cw, const float* __restrict__ cb,
                      float* __restrict__ outf, __bf16* __restrict__ outb)
{
    size_t idx = (size_t)blockIdx.x * 256 + threadIdx.x;   // over ML * D_INNER
    int c  = (int)(idx % D_INNER);
    size_t ml = idx / D_INNER;
    int l = (int)(ml % L_DIM);
    float acc = cb[c];
    #pragma unroll
    for (int j = 0; j < 4; ++j) {
        int ll = l - 3 + j;
        if (ll >= 0)
            acc += cw[c * 4 + j] * xz[(ml - 3 + j) * (2 * D_INNER) + c];
    }
    float s = acc / (1.f + expf(-acc));   // SiLU
    outf[idx] = s;
    outb[idx] = (__bf16)s;
}

// ---------------------------------------------------------------------------
// Chunked first-order recurrence (state = state*dec + u; dec==1 => cumsum)
// ---------------------------------------------------------------------------
template<bool HASDEC>
__global__ __launch_bounds__(256)
void scan_pass1(const float* __restrict__ U, const float* __restrict__ DEC,
                float* __restrict__ Pc, float* __restrict__ Sc)
{
    size_t idx = (size_t)blockIdx.x * 256 + threadIdx.x;   // over B*NC*D
    int d  = (int)(idx % D_INNER);
    int ch = (int)((idx / D_INNER) % NC);
    int b  = (int)(idx / ((size_t)D_INNER * NC));
    size_t base = ((size_t)b * L_DIM + (size_t)ch * CH) * D_INNER + d;
    float P = 1.f, S = 0.f;
    for (int i = 0; i < CH; ++i) {
        float u = U[base + (size_t)i * D_INNER];
        if (HASDEC) {
            float dc = DEC[base + (size_t)i * D_INNER];
            S = S * dc + u;  P *= dc;
        } else {
            S += u;
        }
    }
    size_t o = ((size_t)b * NC + ch) * D_INNER + d;
    Pc[o] = P;  Sc[o] = S;
}

__global__ __launch_bounds__(256)
void scan_pass2(const float* __restrict__ Pc, const float* __restrict__ Sc,
                float* __restrict__ carry, int hasdec)
{
    size_t idx = (size_t)blockIdx.x * 256 + threadIdx.x;   // over B*D
    int d = (int)(idx % D_INNER);
    int b = (int)(idx / D_INNER);
    float c = 0.f;
    for (int ch = 0; ch < NC; ++ch) {
        size_t o = ((size_t)b * NC + ch) * D_INNER + d;
        carry[o] = c;
        c = hasdec ? (c * Pc[o] + Sc[o]) : (c + Sc[o]);
    }
}

template<bool HASDEC, bool OBF>
__global__ __launch_bounds__(256)
void scan_pass3(const float* __restrict__ U, const float* __restrict__ DEC,
                const float* __restrict__ carry,
                float* __restrict__ OUTF, __bf16* __restrict__ OUTB)
{
    size_t idx = (size_t)blockIdx.x * 256 + threadIdx.x;   // over B*NC*D
    int d  = (int)(idx % D_INNER);
    int ch = (int)((idx / D_INNER) % NC);
    int b  = (int)(idx / ((size_t)D_INNER * NC));
    size_t base = ((size_t)b * L_DIM + (size_t)ch * CH) * D_INNER + d;
    float st = carry[((size_t)b * NC + ch) * D_INNER + d];
    for (int i = 0; i < CH; ++i) {
        size_t o = base + (size_t)i * D_INNER;
        float u = U[o];
        if (HASDEC) st = st * DEC[o] + u;
        else        st = st + u;
        if (OBF) OUTB[o] = (__bf16)st;
        else     OUTF[o] = st;
    }
}

// ---------------------------------------------------------------------------
// Gate: y = (scan + x3 * D_ss) * silu(z) -> bf16 (feeds final GEMM only)
// ---------------------------------------------------------------------------
__global__ __launch_bounds__(256)
void gate_kernel(const float* __restrict__ scan_s, const float* __restrict__ x3,
                 const float* __restrict__ Dss, const float* __restrict__ xz,
                 __bf16* __restrict__ y)
{
    size_t idx = (size_t)blockIdx.x * 256 + threadIdx.x;   // over ML * D_INNER
    int d  = (int)(idx % D_INNER);
    size_t ml = idx / D_INNER;
    float z = xz[ml * (2 * D_INNER) + D_INNER + d];
    float xv = scan_s[idx] + x3[idx] * Dss[d];
    y[idx] = (__bf16)(xv * (z / (1.f + expf(-z))));
}

// ---------------------------------------------------------------------------
extern "C" void kernel_launch(void* const* d_in, const int* in_sizes, int n_in,
                              void* d_out, int out_size, void* d_ws, size_t ws_size,
                              hipStream_t stream)
{
    const float* hidden  = (const float*)d_in[0];
    const float* delta   = (const float*)d_in[1];
    const float* in_proj = (const float*)d_in[2];
    const float* conv_w  = (const float*)d_in[3];
    const float* conv_b  = (const float*)d_in[4];
    const float* A_real  = (const float*)d_in[5];
    const float* B_w     = (const float*)d_in[6];
    const float* C_w     = (const float*)d_in[7];
    const float* D_w     = (const float*)d_in[8];
    const float* dt_w    = (const float*)d_in[9];
    const float* dt_b    = (const float*)d_in[10];
    const float* A_log   = (const float*)d_in[11];
    const float* D_ss    = (const float*)d_in[12];
    const float* out_w   = (const float*)d_in[13];
    float* out = (float*)d_out;

    // ---- workspace layout -------------------------------------------------
    const size_t NE = ML * (size_t)D_INNER;                // 16.7M
    float* ws   = (float*)d_ws;
    float* xz   = ws;                                      // ML*4096
    float* fA   = xz + ML * (size_t)(2 * D_INNER);         // xs, later decay
    float* fB   = fA + NE;                                 // x2, later scan out
    float* fC   = fB + NE;                                 // Bx, later x3
    float* Pc    = fC + NE;
    float* Sc    = Pc + (size_t)B_DIM * NC * D_INNER;
    float* carry = Sc + (size_t)B_DIM * NC * D_INNER;
    __bf16* bws   = (__bf16*)(carry + (size_t)B_DIM * NC * D_INNER);
    __bf16* h16   = bws;                                   // ML*1024
    __bf16* d16   = h16  + ML * (size_t)D_MODEL;           // ML*64
    __bf16* wIn16 = d16  + ML * (size_t)DT_RANK;
    __bf16* wA16  = wIn16 + (size_t)2 * D_INNER * D_MODEL;
    __bf16* wB16  = wA16  + (size_t)D_INNER * D_INNER;
    __bf16* wC16  = wB16  + (size_t)D_INNER * D_INNER;
    __bf16* wD16  = wC16  + (size_t)D_INNER * D_INNER;
    __bf16* wDt16 = wD16  + (size_t)D_INNER * D_INNER;
    __bf16* wOut16= wDt16 + (size_t)D_INNER * DT_RANK;
    __bf16* xs16  = wOut16 + (size_t)D_MODEL * D_INNER;
    __bf16* x216  = xs16 + NE;
    __bf16* st16  = x216 + NE;
    __bf16* y16   = st16 + NE;

    const dim3 blk(256);
    const int  MBLK = (int)(ML / BM);                      // 128
    const int  eBlocks = (int)(NE / 256);
    const int  sBlocks = (int)(((size_t)B_DIM * NC * D_INNER) / 256);
    const int  cBlocks = (int)(((size_t)B_DIM * D_INNER) / 256);
    auto cvt = [&](const float* src, __bf16* dst, size_t n) {
        cvt_bf16_kernel<<<(unsigned)(n / 1024), blk, 0, stream>>>(src, dst);
    };

    // 0) bf16 conversions of all GEMM operands
    cvt(hidden,  h16,   ML * (size_t)D_MODEL);
    cvt(delta,   d16,   ML * (size_t)DT_RANK);
    cvt(in_proj, wIn16, (size_t)2 * D_INNER * D_MODEL);
    cvt(A_real,  wA16,  (size_t)D_INNER * D_INNER);
    cvt(B_w,     wB16,  (size_t)D_INNER * D_INNER);
    cvt(C_w,     wC16,  (size_t)D_INNER * D_INNER);
    cvt(D_w,     wD16,  (size_t)D_INNER * D_INNER);
    cvt(dt_w,    wDt16, (size_t)D_INNER * DT_RANK);
    cvt(out_w,   wOut16,(size_t)D_MODEL * D_INNER);

    // 1) xz = hidden @ in_proj^T            (M=8192, K=1024, N=4096)
    wmma_gemm<true, false, 0, false><<<dim3(2 * D_INNER / BN, MBLK), blk, 0, stream>>>(
        h16, wIn16, nullptr, nullptr, nullptr, nullptr, nullptr,
        xz, nullptr, 2 * D_INNER, D_MODEL);

    // 2) xs = silu(depthwise_conv(x) + b)
    conv_silu_kernel<<<eBlocks, blk, 0, stream>>>(xz, conv_w, conv_b, fA, xs16);

    // 3) x2 = xs + xs @ A_real              (NN weight)
    wmma_gemm<false, false, 1, true><<<dim3(D_INNER / BN, MBLK), blk, 0, stream>>>(
        xs16, wA16, nullptr, nullptr, fA, nullptr, nullptr,
        fB, x216, D_INNER, D_INNER);

    // 4) Bx = x2 @ B_w^T
    wmma_gemm<true, false, 0, false><<<dim3(D_INNER / BN, MBLK), blk, 0, stream>>>(
        x216, wB16, nullptr, nullptr, nullptr, nullptr, nullptr,
        fC, nullptr, D_INNER, D_INNER);

    // 5) states = cumsum_L(Bx) -> bf16
    scan_pass1<false><<<sBlocks, blk, 0, stream>>>(fC, nullptr, Pc, Sc);
    scan_pass2<<<cBlocks, blk, 0, stream>>>(Pc, Sc, carry, 0);
    scan_pass3<false, true><<<sBlocks, blk, 0, stream>>>(fC, nullptr, carry, nullptr, st16);

    // 6) x3 = x2 + states @ C_w^T + x2 @ D_w^T   (dual product)
    wmma_gemm<true, true, 1, false><<<dim3(D_INNER / BN, MBLK), blk, 0, stream>>>(
        st16, wC16, x216, wD16, fB, nullptr, nullptr,
        fC, nullptr, D_INNER, D_INNER);

    // 7) decay = exp(-exp(A_log) * softplus(delta @ dt_w^T + dt_b))
    wmma_gemm<true, false, 2, false><<<dim3(D_INNER / BN, MBLK), blk, 0, stream>>>(
        d16, wDt16, nullptr, nullptr, nullptr, dt_b, A_log,
        fA, nullptr, D_INNER, DT_RANK);

    // 8) selective scan: state = state*decay + x3
    scan_pass1<true><<<sBlocks, blk, 0, stream>>>(fC, fA, Pc, Sc);
    scan_pass2<<<cBlocks, blk, 0, stream>>>(Pc, Sc, carry, 1);
    scan_pass3<true, false><<<sBlocks, blk, 0, stream>>>(fC, fA, carry, fB, nullptr);

    // 9) y = (scan + x3*D_ss) * silu(z) -> bf16
    gate_kernel<<<eBlocks, blk, 0, stream>>>(fB, fC, D_ss, xz, y16);

    // 10) out = y @ out_proj^T              (N=1024)
    wmma_gemm<true, false, 0, false><<<dim3(D_MODEL / BN, MBLK), blk, 0, stream>>>(
        y16, wOut16, nullptr, nullptr, nullptr, nullptr, nullptr,
        out, nullptr, D_MODEL, D_INNER);
}
